// GraphRec_88630945120921
// MI455X (gfx1250) — matrix operand order repaired and checked by reading, not compiled, not run on hardware
//
#include <hip/hip_runtime.h>

typedef unsigned int   u32;
typedef unsigned short u16;
typedef __attribute__((ext_vector_type(16))) __bf16 v16bf;
typedef __attribute__((ext_vector_type(8)))  float  v8f;

__device__ __forceinline__ u16 f2bf(float f) {
  union { float f; u32 u; } v; v.f = f;
  u32 r = v.u + 0x7FFFu + ((v.u >> 16) & 1u);
  return (u16)(r >> 16);
}
__device__ __forceinline__ float bf2f(u16 h) {
  union { u32 u; float f; } v; v.u = ((u32)h) << 16;
  return v.f;
}

union FragU { u32 u[8]; v16bf v; };

// ---------------------------------------------------------------------------
// Fused gather + concat + f32->bf16 row builder.
// out[m, 0:64]  = left  (hist-gather from f32 table, or dense bf16 copy)
// out[m,64:128] = right (hist-gather from f32 table, or node-gather rep)
// ---------------------------------------------------------------------------
__global__ void k_build_rows(u16* __restrict__ out, int M, int L,
                             const int* __restrict__ nodes,
                             const int* __restrict__ left_hist,
                             const float* __restrict__ left_tbl,
                             const u16* __restrict__ left_dense,
                             const int* __restrict__ right_hist,
                             const float* __restrict__ right_tbl) {
  int g = blockIdx.x * blockDim.x + threadIdx.x;   // dword id over [M][64]
  int m = g >> 6, d = g & 63;
  if (m >= M) return;
  int b = m / L, l = m - b * L;
  int node = nodes[b];
  u32 val;
  if (d < 32) {
    int c = d * 2;
    if (left_hist) {
      int idx = left_hist[(size_t)node * L + l];
      float2 f = *(const float2*)(left_tbl + (size_t)idx * 64 + c);
      val = (u32)f2bf(f.x) | ((u32)f2bf(f.y) << 16);
    } else {
      val = *(const u32*)(left_dense + (size_t)m * 64 + c);
    }
  } else {
    int c = (d - 32) * 2;
    int idx = right_hist ? right_hist[(size_t)node * L + l] : node;
    float2 f = *(const float2*)(right_tbl + (size_t)idx * 64 + c);
    val = (u32)f2bf(f.x) | ((u32)f2bf(f.y) << 16);
  }
  *(u32*)(out + (size_t)m * 128 + d * 2) = val;
}

// ---------------------------------------------------------------------------
// WMMA GEMM: Y[M,N](bf16, ld=ldy) = act( (X[M,K]bf16 @ W[K,N]f32 + db) * s + bb )
// 256 threads = 8 waves; wave owns 16 rows; N/16 tiles of v_wmma_f32_16x16x32_bf16.
// A frags straight from global (lane=M, halves K+0/+8/+16).
// B frags from LDS weights stored FRAGMENT-MAJOR: for fragment f=(kk/32)*NT+nt,
// lane k's 16 bf16 values are contiguous 32B -> two ds_load_b128, no addr math,
// conflict-free (consecutive lanes hit consecutive 256-bit chunks).
// ---------------------------------------------------------------------------
template <int K, int N>
__global__ __launch_bounds__(256) void k_gemm(const u16* __restrict__ X,
                                              const float* __restrict__ W,
                                              const float* __restrict__ db,
                                              const float* __restrict__ bs,
                                              const float* __restrict__ bb,
                                              u16* __restrict__ Y,
                                              int ldy, int relu) {
  constexpr int NT = N / 16;
  constexpr int NP = N + 2;           // pad for the output staging tile only
  __shared__ u16 Wb[K * N];           // fragment-major weight copy (bf16)
  __shared__ u16 St[8 * 16 * NP];

  const int tid = threadIdx.x;
  const int lane = tid & 31, wv = tid >> 5;

  // Fill Wb: i = ((f*32 + ln)*16 + j); f=(kk/32)*NT+nt; value = W[(kk + 8*(ln/16) + 16*(j/8) + j%8)*N + nt*16 + ln%16]
  for (int i = tid; i < K * N; i += 256) {
    int f  = i >> 9;
    int r  = i & 511;
    int ln = r >> 4;
    int j  = r & 15;
    int kk = (f / NT) * 32;
    int nt = f - (f / NT) * NT;
    Wb[i] = f2bf(W[(kk + (ln >> 4) * 8 + (j >> 3) * 16 + (j & 7)) * N + nt * 16 + (ln & 15)]);
  }
  __syncthreads();

  const int half = lane >> 4, l15 = lane & 15;
  const int rowbase = blockIdx.x * 128 + wv * 16;
  const u32* Xr = (const u32*)(X + (size_t)(rowbase + l15) * K);

  v8f acc[NT];
#pragma unroll
  for (int nt = 0; nt < NT; ++nt) acc[nt] = (v8f)(0.0f);

#pragma unroll
  for (int kk = 0; kk < K; kk += 32) {
    const int kb = kk + half * 8;
    FragU a;
    uint4 t0 = *(const uint4*)(Xr + (kb >> 1));
    uint4 t1 = *(const uint4*)(Xr + ((kb + 16) >> 1));
    a.u[0] = t0.x; a.u[1] = t0.y; a.u[2] = t0.z; a.u[3] = t0.w;
    a.u[4] = t1.x; a.u[5] = t1.y; a.u[6] = t1.z; a.u[7] = t1.w;
#pragma unroll
    for (int nt = 0; nt < NT; ++nt) {
      FragU bf;
      const uint4* wp =
          (const uint4*)(Wb + ((((kk >> 5) * NT + nt) * 32 + lane) << 4));
      uint4 w0 = wp[0], w1 = wp[1];
      bf.u[0] = w0.x; bf.u[1] = w0.y; bf.u[2] = w0.z; bf.u[3] = w0.w;
      bf.u[4] = w1.x; bf.u[5] = w1.y; bf.u[6] = w1.z; bf.u[7] = w1.w;
      acc[nt] = __builtin_amdgcn_wmma_f32_16x16x32_bf16(
          false, a.v, false, bf.v, (short)0, acc[nt], false, false);
    }
  }

  // Epilogue: bias + folded inference-BN (gamma/sqrt(1+eps)) + ReLU -> LDS stage
  const float RSQ = 0.99950037f;      // 1/sqrt(1+1e-3)
#pragma unroll
  for (int nt = 0; nt < NT; ++nt) {
    int n = nt * 16 + l15;
    float dbv = db ? db[n] : 0.0f;
    float sv  = bs ? bs[n] * RSQ : 1.0f;
    float bbv = bb ? bb[n] : 0.0f;
#pragma unroll
    for (int r = 0; r < 8; ++r) {
      float v = (acc[nt][r] + dbv) * sv + bbv;
      if (relu) v = fmaxf(v, 0.0f);
      St[(wv * 16 + r + half * 8) * NP + n] = f2bf(v);
    }
  }
  // Coalesced bf16 store of the wave-private 16xN tile
  const int ndw = N >> 1;
  for (int i = lane; i < 16 * ndw; i += 32) {
    int r = i / ndw, c = i - r * ndw;
    u32 d = *(const u32*)(St + (wv * 16 + r) * NP + 2 * c);
    *(u32*)(Y + (size_t)(rowbase + r) * ldy + 2 * c) = d;
  }
}

// ---------------------------------------------------------------------------
// N=1 projection: out[m] = X[m,:K].bf16 . w + b
// ---------------------------------------------------------------------------
__global__ void k_dot(const u16* __restrict__ X, const float* __restrict__ w,
                      const float* __restrict__ bias, float* __restrict__ out,
                      int M, int K) {
  int m = blockIdx.x * blockDim.x + threadIdx.x;
  if (m >= M) return;
  const u32* xr = (const u32*)(X + (size_t)m * K);
  float acc = bias[0];
  for (int j = 0; j < (K >> 1); ++j) {
    u32 u = xr[j];
    acc += bf2f((u16)u) * w[2 * j] + bf2f((u16)(u >> 16)) * w[2 * j + 1];
  }
  out[m] = acc;
}

// ---------------------------------------------------------------------------
// Per-b softmax over L logits + weighted sum of o rows; writes concat buffer:
// outcat[b,64:128] = sum_l softmax_l * o[b,l,:]; outcat[b,0:64] = left half
// (rep gather from f32 table, or dense bf16 rows). 64 threads per block.
// ---------------------------------------------------------------------------
__global__ void k_softmax_wsum(const float* __restrict__ logits,
                               const u16* __restrict__ O, int ostride,
                               u16* __restrict__ outcat,
                               const int* __restrict__ nodes,
                               const float* __restrict__ left_tbl,
                               const u16* __restrict__ left_dense, int L) {
  int b = blockIdx.x, t = threadIdx.x;
  __shared__ float lg[64];
  __shared__ float ew[64];
  if (t < L) lg[t] = logits[(size_t)b * L + t];
  __syncthreads();
  float mx = -1e30f;
  for (int l = 0; l < L; ++l) mx = fmaxf(mx, lg[l]);
  if (t < L) ew[t] = __expf(lg[t] - mx);
  __syncthreads();
  float s = 0.0f;
  for (int l = 0; l < L; ++l) s += ew[l];
  float inv = 1.0f / s;
  float acc = 0.0f;
  for (int l = 0; l < L; ++l)
    acc += ew[l] * bf2f(O[((size_t)b * L + l) * ostride + t]);
  acc *= inv;
  outcat[(size_t)b * 128 + 64 + t] = f2bf(acc);
  float lv = left_tbl ? left_tbl[(size_t)nodes[b] * 64 + t]
                      : bf2f(left_dense[(size_t)b * 64 + t]);
  outcat[(size_t)b * 128 + t] = f2bf(lv);
}

// ---------------------------------------------------------------------------
extern "C" void kernel_launch(void* const* d_in, const int* in_sizes, int n_in,
                              void* d_out, int out_size, void* d_ws, size_t ws_size,
                              hipStream_t stream) {
  const int B = 8192, L = 50;
  const int ML = B * L;  // 409600, divisible by 128/256

  const int*   nodes_u   = (const int*)d_in[0];
  const int*   nodes_v   = (const int*)d_in[1];
  const int*   hist_u_i  = (const int*)d_in[2];
  const int*   hist_u_r  = (const int*)d_in[3];
  const int*   hist_v_u  = (const int*)d_in[4];
  const int*   hist_v_r  = (const int*)d_in[5];
  const int*   soc_adj   = (const int*)d_in[6];
  const float* u2e = (const float*)d_in[7];
  const float* v2e = (const float*)d_in[8];
  const float* r2e = (const float*)d_in[9];
#define FW(i) ((const float*)d_in[(i)])
  // weights: wr1u 10/11 wr2u 12/13 attu1 14/15 attu2 16/17 attu3 18/19 encu 20/21
  //          wr1v 22/23 wr2v 24/25 attv1 26/27 attv2 28/29 attv3 30/31 encv 32/33
  //          atts1 34/35 atts2 36/37 atts3 38/39 encs 40/41
  //          wur1 42/43 wur2 44/45 wvr1 46/47 wvr2 48/49
  //          wuv1 50/51 wuv2 52/53 wuv3 54/55  bn: 56..63

  char* ws = (char*)d_ws;
  size_t off = 0;
  auto alloc = [&](size_t bytes) -> char* {
    char* p = ws + off;
    off += (bytes + 255) & ~(size_t)255;
    return p;
  };
  u16* P0 = (u16*)alloc((size_t)ML * 128 * 2);  // E rows / A rows (reused)
  u16* P1 = (u16*)alloc((size_t)ML * 64 * 2);   // H1 / T1
  u16* P2 = (u16*)alloc((size_t)ML * 64 * 2);   // O (alive through wsum)
  u16* P3 = (u16*)alloc((size_t)ML * 64 * 2);   // T2
  float* LG = (float*)alloc((size_t)ML * 4);    // logits
  u16* Cu  = (u16*)alloc((size_t)B * 128 * 2);
  u16* Hu  = (u16*)alloc((size_t)B * 64 * 2);
  u16* Cs  = (u16*)alloc((size_t)B * 128 * 2);
  u16* EU  = (u16*)alloc((size_t)B * 64 * 2);
  u16* Cv  = (u16*)alloc((size_t)B * 128 * 2);
  u16* EV  = (u16*)alloc((size_t)B * 64 * 2);
  u16* Y1  = (u16*)alloc((size_t)B * 64 * 2);
  u16* Cuv = (u16*)alloc((size_t)B * 128 * 2);
  u16* Z1  = (u16*)alloc((size_t)B * 64 * 2);
  u16* Z2  = (u16*)alloc((size_t)B * 16 * 2);
  (void)ws_size; (void)in_sizes; (void)n_in; (void)out_size;

  const int BR = (ML * 64) / 256;  // build_rows grid for M=ML
  const dim3 T256(256), T64(64);

  // ---------------- user side ----------------
  k_build_rows<<<BR, T256, 0, stream>>>(P0, ML, L, nodes_u, hist_u_i, v2e, nullptr, hist_u_r, r2e);
  k_gemm<128, 64><<<ML / 128, T256, 0, stream>>>(P0, FW(10), FW(11), nullptr, nullptr, P1, 64, 1);
  k_gemm<64, 64><<<ML / 128, T256, 0, stream>>>(P1, FW(12), FW(13), nullptr, nullptr, P2, 64, 1);
  k_build_rows<<<BR, T256, 0, stream>>>(P0, ML, L, nodes_u, nullptr, nullptr, P2, nullptr, u2e);
  k_gemm<128, 64><<<ML / 128, T256, 0, stream>>>(P0, FW(14), FW(15), nullptr, nullptr, P1, 64, 1);
  k_gemm<64, 64><<<ML / 128, T256, 0, stream>>>(P1, FW(16), FW(17), nullptr, nullptr, P3, 64, 1);
  k_dot<<<ML / 256, T256, 0, stream>>>(P3, FW(18), FW(19), LG, ML, 64);
  k_softmax_wsum<<<B, T64, 0, stream>>>(LG, P2, 64, Cu, nodes_u, u2e, nullptr, L);
  k_gemm<128, 64><<<B / 128, T256, 0, stream>>>(Cu, FW(20), FW(21), nullptr, nullptr, Hu, 64, 1);

  // ---------------- social side ----------------
  k_build_rows<<<BR, T256, 0, stream>>>(P0, ML, L, nodes_u, soc_adj, u2e, nullptr, nullptr, u2e);
  k_gemm<128, 64><<<ML / 128, T256, 0, stream>>>(P0, FW(34), FW(35), nullptr, nullptr, P1, 64, 1);
  k_gemm<64, 64><<<ML / 128, T256, 0, stream>>>(P1, FW(36), FW(37), nullptr, nullptr, P3, 64, 1);
  k_dot<<<ML / 256, T256, 0, stream>>>(P3, FW(38), FW(39), LG, ML, 64);
  k_softmax_wsum<<<B, T64, 0, stream>>>(LG, P0, 128, Cs, nodes_u, nullptr, Hu, L);
  k_gemm<128, 64><<<B / 128, T256, 0, stream>>>(Cs, FW(40), FW(41), nullptr, nullptr, EU, 64, 1);

  // ---------------- item side ----------------
  k_build_rows<<<BR, T256, 0, stream>>>(P0, ML, L, nodes_v, hist_v_u, u2e, nullptr, hist_v_r, r2e);
  k_gemm<128, 64><<<ML / 128, T256, 0, stream>>>(P0, FW(22), FW(23), nullptr, nullptr, P1, 64, 1);
  k_gemm<64, 64><<<ML / 128, T256, 0, stream>>>(P1, FW(24), FW(25), nullptr, nullptr, P2, 64, 1);
  k_build_rows<<<BR, T256, 0, stream>>>(P0, ML, L, nodes_v, nullptr, nullptr, P2, nullptr, v2e);
  k_gemm<128, 64><<<ML / 128, T256, 0, stream>>>(P0, FW(26), FW(27), nullptr, nullptr, P1, 64, 1);
  k_gemm<64, 64><<<ML / 128, T256, 0, stream>>>(P1, FW(28), FW(29), nullptr, nullptr, P3, 64, 1);
  k_dot<<<ML / 256, T256, 0, stream>>>(P3, FW(30), FW(31), LG, ML, 64);
  k_softmax_wsum<<<B, T64, 0, stream>>>(LG, P2, 64, Cv, nodes_v, v2e, nullptr, L);
  k_gemm<128, 64><<<B / 128, T256, 0, stream>>>(Cv, FW(32), FW(33), nullptr, nullptr, EV, 64, 1);

  // ---------------- final MLP ----------------
  // x_u = (relu(bn1(EU@wur1+b))) @ wur2 + b  -> Cuv[:, 0:64]
  k_gemm<64, 64><<<B / 128, T256, 0, stream>>>(EU, FW(42), FW(43), FW(56), FW(57), Y1, 64, 1);
  k_gemm<64, 64><<<B / 128, T256, 0, stream>>>(Y1, FW(44), FW(45), nullptr, nullptr, Cuv, 128, 0);
  // x_v -> Cuv[:, 64:128]
  k_gemm<64, 64><<<B / 128, T256, 0, stream>>>(EV, FW(46), FW(47), FW(58), FW(59), Y1, 64, 1);
  k_gemm<64, 64><<<B / 128, T256, 0, stream>>>(Y1, FW(48), FW(49), nullptr, nullptr, Cuv + 64, 128, 0);
  // x = relu(bn3(Cuv@wuv1+b)) ; x = relu(bn4(x@wuv2+b)) ; out = x@wuv3+b
  k_gemm<128, 64><<<B / 128, T256, 0, stream>>>(Cuv, FW(50), FW(51), FW(60), FW(61), Z1, 64, 1);
  k_gemm<64, 16><<<B / 128, T256, 0, stream>>>(Z1, FW(52), FW(53), FW(62), FW(63), Z2, 16, 1);
  k_dot<<<B / 256, T256, 0, stream>>>(Z2, FW(54), FW(55), (float*)d_out, B, 16);
#undef FW
}
